// TorchKAN_17643725652391
// MI455X (gfx1250) — matrix operand order repaired and checked by reading, not compiled, not run on hardware
//
#include <hip/hip_runtime.h>

// KAN 2-layer: out = kan(kan(x, w1, a1), w2, a2)
// v_wmma_f32_16x16x32_f16 GEMMs, f32 accumulate. Pre-pass converts operands
// to f16 in d_ws. Main loop: double-buffered LDS, CDNA5 async global->LDS
// copies (GLOBAL_LOAD_ASYNC_TO_LDS_B128, ASYNCcnt) when available.
// d_ws layout (halves): xh[8388608] | w1h[262144] | w2h[262144] | h[33554432]
// => ~85MB workspace required.

typedef __attribute__((ext_vector_type(16))) _Float16 v16h;
typedef __attribute__((ext_vector_type(8)))  _Float16 v8h;
typedef __attribute__((ext_vector_type(8)))  float    v8f;
typedef __attribute__((ext_vector_type(4)))  float    v4f;
typedef __attribute__((ext_vector_type(4)))  int      v4i;

#define BM 128
#define BN 128
#define BK 32
#define LDS_STRIDE 40   // halves/row: 80B, 16B-aligned, conflict-free row gather

#if defined(__gfx1250__) && __has_builtin(__builtin_amdgcn_global_load_async_to_lds_b128)
#define KAN_ASYNC 1
#else
#define KAN_ASYNC 0
#endif

#define AS1 __attribute__((address_space(1)))
#define AS3 __attribute__((address_space(3)))

// One 16-byte global->LDS copy (async on CDNA5, VGPR-staged fallback).
__device__ __forceinline__ void cp16_g2l(const _Float16* __restrict__ g,
                                         _Float16* l) {
#if KAN_ASYNC
  // Builtin signature (from toolchain diagnostic): (v4i AS1* src, v4i AS3* dst,
  // imm offset, imm cpol). C-style cast drops const + changes addrspace.
  __builtin_amdgcn_global_load_async_to_lds_b128((AS1 v4i*)g, (AS3 v4i*)l, 0, 0);
#else
  *(v8h*)l = *(const v8h*)g;
#endif
}

template <int N>
__device__ __forceinline__ void wait_async() {
#if KAN_ASYNC
#if __has_builtin(__builtin_amdgcn_s_wait_asynccnt)
  __builtin_amdgcn_s_wait_asynccnt(N);
#else
  asm volatile("s_wait_asynccnt %0" ::"i"(N) : "memory");
#endif
#endif
}

// ---------------- f32 -> f16 pre-pass (8 elements/thread) ----------------
__global__ __launch_bounds__(256) void cvt_f32_f16(const float* __restrict__ in,
                                                   _Float16* __restrict__ out,
                                                   int n8) {
  int i = blockIdx.x * 256 + threadIdx.x;
  if (i < n8) {
    v4f a = ((const v4f*)in)[2 * i];
    v4f b = ((const v4f*)in)[2 * i + 1];
    v8h o;
#pragma unroll
    for (int j = 0; j < 4; ++j) { o[j] = (_Float16)a[j]; o[j + 4] = (_Float16)b[j]; }
    ((v8h*)out)[i] = o;
  }
}

// Assemble a WMMA fragment from two contiguous 8-half groups in LDS.
__device__ __forceinline__ v16h ld_frag(const _Float16* __restrict__ p, int o0, int o1) {
  v8h lo = *(const v8h*)(p + o0);
  v8h hi = *(const v8h*)(p + o1);
  v16h r;
#pragma unroll
  for (int i = 0; i < 8; ++i) { r[i] = lo[i]; r[i + 8] = hi[i]; }
  return r;
}

// Branchless 6-activation blend: one v_exp, one v_rcp (shared denominator),
// one v_log. sigmoid = r1 | 1-r1; tanh = +/-(2*r2-1);
// softplus = relu + ln2*log2(1+em); leaky folded into coefficients.
__device__ __forceinline__ float kan_blend(float s, float c_sig, float c_th,
                                           float c_re, float c_s, float c_lg,
                                           float c_el) {
  float em  = __builtin_amdgcn_exp2f(-1.4426950408889634f * fabsf(s));
  float p   = 1.0f + em;
  float em2 = em * em;
  float q   = 1.0f + em2;
  float rD  = __builtin_amdgcn_rcpf(p * q);
  float r1  = q * rD;                     // 1/(1+em)
  float r2  = p * rD;                     // 1/(1+em2)
  bool  pos = s >= 0.0f;
  float sig = pos ? r1 : (1.0f - r1);
  float tha = fmaf(2.0f, r2, -1.0f);      // tanh(|s|)
  float th  = pos ? tha : -tha;
  float re  = fmaxf(s, 0.0f);
  float lg2 = __builtin_amdgcn_logf(p);   // log2(1+em)
  float el  = pos ? s : (em - 1.0f);      // elu (em == e^s for s<=0)
  float o = c_sig * sig;
  o = fmaf(c_th, th, o);
  o = fmaf(c_re, re, o);
  o = fmaf(c_s, s, o);
  o = fmaf(c_lg, lg2, o);
  o = fmaf(c_el, el, o);
  return o;
}

// ---------------- fused GEMM(+activation blend) kernel ----------------
template <typename TOUT>
__global__ __launch_bounds__(256) void kan_wmma_kernel(
    const _Float16* __restrict__ X,  // [B, K] row-major f16
    const _Float16* __restrict__ W,  // [N, K] row-major f16
    const float* __restrict__ A,     // [N, 6] f32
    TOUT* __restrict__ Out,          // [B, N] row-major
    int K, int N) {
  __shared__ _Float16 xs[2][BM * LDS_STRIDE];
  __shared__ _Float16 wl[2][BN * LDS_STRIDE];

  const int tid  = threadIdx.x;
  const int lane = tid & 31;
  const int wave = tid >> 5;
  const int wM   = (wave & 1) * 64;
  const int wN   = (wave >> 1) * 32;
  const int g    = lane >> 4;
  const int l16  = lane & 15;
  const int M0   = blockIdx.x * BM;
  const int N0   = blockIdx.y * BN;

  // Per-thread copy coords: 512 8-half chunks per 128x32 tile, 2/thread.
  const int q0 = tid, q1 = tid + 256;
  const int r0 = q0 >> 2, cc0 = (q0 & 3) * 8;
  const int r1 = q1 >> 2, cc1 = (q1 & 3) * 8;
  const _Float16* xg0 = X + (size_t)(M0 + r0) * K + cc0;
  const _Float16* xg1 = X + (size_t)(M0 + r1) * K + cc1;
  const _Float16* wg0 = W + (size_t)(N0 + r0) * K + cc0;
  const _Float16* wg1 = W + (size_t)(N0 + r1) * K + cc1;
  const int lo0 = r0 * LDS_STRIDE + cc0;
  const int lo1 = r1 * LDS_STRIDE + cc1;

  v8f acc[4][2];
#pragma unroll
  for (int mi = 0; mi < 4; ++mi)
#pragma unroll
    for (int ni = 0; ni < 2; ++ni) acc[mi][ni] = (v8f){};

  const int T = K / BK;

  // Issue tile 0 into buffer 0 (4 async copies per thread).
  cp16_g2l(xg0, &xs[0][lo0]);
  cp16_g2l(xg1, &xs[0][lo1]);
  cp16_g2l(wg0, &wl[0][lo0]);
  cp16_g2l(wg1, &wl[0][lo1]);

  for (int kt = 0; kt < T; ++kt) {
    const int b0 = kt & 1;
    __syncthreads();  // all waves done reading buf[1-b0] from previous compute

    if (kt + 1 < T) {
      // Issue next tile into the other buffer; overlaps with this tile's WMMAs.
      const int kn = (kt + 1) * BK;
      cp16_g2l(xg0 + kn, &xs[1 - b0][lo0]);
      cp16_g2l(xg1 + kn, &xs[1 - b0][lo1]);
      cp16_g2l(wg0 + kn, &wl[1 - b0][lo0]);
      cp16_g2l(wg1 + kn, &wl[1 - b0][lo1]);
      wait_async<4>();  // first 4 outstanding (tile kt) have landed, in order
    } else {
      wait_async<0>();
    }
    __syncthreads();  // every wave waited its own ASYNCcnt -> tile kt in LDS

    const _Float16* xb = xs[b0];
    const _Float16* wb = wl[b0];
    v16h bf[2];
#pragma unroll
    for (int ni = 0; ni < 2; ++ni) {
      int n = wN + ni * 16 + l16;
      bf[ni] = ld_frag(wb + n * LDS_STRIDE, g * 16, g * 16 + 8);
    }
#pragma unroll
    for (int mi = 0; mi < 4; ++mi) {
      int r = wM + mi * 16 + l16;
      v16h af = ld_frag(xb + r * LDS_STRIDE, g * 16, g * 16 + 8);
#pragma unroll
      for (int ni = 0; ni < 2; ++ni)
        acc[mi][ni] = __builtin_amdgcn_wmma_f32_16x16x32_f16(
            false, af, false, bf[ni], (short)0, acc[mi][ni], false, false);
    }
  }

  // Epilogue: blend + store.
#pragma unroll
  for (int ni = 0; ni < 2; ++ni) {
    int n = N0 + wN + ni * 16 + l16;
    const float* av = A + (size_t)n * 6;
    float a0 = av[0], a1 = av[1], a2 = av[2], a3 = av[3], a4 = av[4], a5 = av[5];
    float c_re = a2 + 0.99f * a3 + a4;
    float c_s  = 0.01f * a3;
    float c_lg = a4 * 0.6931471805599453f;
#pragma unroll
    for (int mi = 0; mi < 4; ++mi) {
#pragma unroll
      for (int r = 0; r < 8; ++r) {
        float s = acc[mi][ni][r];
        float o = kan_blend(s, a0, a1, c_re, c_s, c_lg, a5);
        int row = M0 + wM + mi * 16 + g * 8 + r;  // C/D layout: VGPR r -> M = r+8g
        Out[(size_t)row * N + n] = (TOUT)o;
      }
    }
  }
}

extern "C" void kernel_launch(void* const* d_in, const int* in_sizes, int n_in,
                              void* d_out, int out_size, void* d_ws, size_t ws_size,
                              hipStream_t stream) {
  const float* x  = (const float*)d_in[0];   // [32768, 256]
  const float* w1 = (const float*)d_in[1];   // [1024, 256]
  const float* a1 = (const float*)d_in[2];   // [1024, 6]
  const float* w2 = (const float*)d_in[3];   // [256, 1024]
  const float* a2 = (const float*)d_in[4];   // [256, 6]

  const int B = 32768, D_IN = 256, H = 1024, D_OUT = 256;

  _Float16* xh  = (_Float16*)d_ws;                     // B*D_IN
  _Float16* w1h = xh + (size_t)B * D_IN;               // H*D_IN
  _Float16* w2h = w1h + (size_t)H * D_IN;              // D_OUT*H
  _Float16* h   = w2h + (size_t)D_OUT * H;             // B*H

  // f32 -> f16 pre-pass (memory bound, ~55MB total traffic)
  cvt_f32_f16<<<(B * D_IN / 8 + 255) / 256, 256, 0, stream>>>(x, xh, B * D_IN / 8);
  cvt_f32_f16<<<(H * D_IN / 8 + 255) / 256, 256, 0, stream>>>(w1, w1h, H * D_IN / 8);
  cvt_f32_f16<<<(D_OUT * H / 8 + 255) / 256, 256, 0, stream>>>(w2, w2h, D_OUT * H / 8);

  dim3 block(256);
  kan_wmma_kernel<_Float16>
      <<<dim3(B / BM, H / BN), block, 0, stream>>>(xh, w1h, a1, h, D_IN, H);
  kan_wmma_kernel<float>
      <<<dim3(B / BM, D_OUT / BN), block, 0, stream>>>(h, w2h, a2, (float*)d_out, H, D_OUT);
}
